// Attention_Layer_51771535786510
// MI455X (gfx1250) — matrix operand order repaired and checked by reading, not compiled, not run on hardware
//
#include <hip/hip_runtime.h>

// ---------------------------------------------------------------------------
// MI455X (gfx1250) attention:  out = softmax(Q K^T) V,  Q/K/V = x @ W{q,k,v}^T
// B=8, S=2048, D=384.  f16 WMMA (16x16x32) with f32 accumulation.
// ---------------------------------------------------------------------------

#define DEV __device__ __forceinline__

typedef _Float16 half_t;
typedef __attribute__((ext_vector_type(4)))  _Float16 v4h;
typedef __attribute__((ext_vector_type(8)))  _Float16 v8h;
typedef __attribute__((ext_vector_type(16))) _Float16 v16h;
typedef __attribute__((ext_vector_type(4)))  float    v4f;
typedef __attribute__((ext_vector_type(8)))  float    v8f;

constexpr int Dm   = 384;
constexpr int Sq   = 2048;
constexpr int Bb   = 8;
constexpr int Mrow = Bb * Sq;       // 16384 total rows
constexpr int QSTR = 392;           // padded LDS row stride in halves (16B-aligned, breaks 64-bank cycling)
constexpr int PSTR = 40;            // padded P-tile row stride in halves

// Build a 16-half fragment from two contiguous 16-byte chunks (two b128 loads).
DEV v16h make_frag(const half_t* lo, const half_t* hi) {
  v8h a = *(const v8h*)lo;
  v8h b = *(const v8h*)hi;
  return __builtin_shufflevector(a, b, 0,1,2,3,4,5,6,7,8,9,10,11,12,13,14,15);
}

DEV v8f wmma_f16(v16h a, v16h b, v8f c) {
  // D = A(16x32 f16) x B(32x16 f16) + C(16x16 f32)
  return __builtin_amdgcn_wmma_f32_16x16x32_f16(false, a, false, b, (short)0, c,
                                                false, false);
}

// ---------------------------------------------------------------------------
// Kernel 1: Y = x @ W^T (torch Linear).  A-tile = x rows, B-col n = W row n.
// blockIdx = (m-tile, n-tile of 64, which matrix).  128 threads = 4 waves,
// each wave one 16x16 output sub-tile.  V output is stored TRANSPOSED so the
// attention P*V B-fragments are contiguous.
// ---------------------------------------------------------------------------
__global__ __launch_bounds__(128)
void qkv_proj_kernel(const float* __restrict__ x,
                     const float* __restrict__ Wq,
                     const float* __restrict__ Wk,
                     const float* __restrict__ Wv,
                     half_t* __restrict__ Qh,   // [M, D] row-major f16
                     half_t* __restrict__ Kh,   // [M, D] row-major f16
                     half_t* __restrict__ Vt) { // [B, D, S] f16 (transposed)
  __shared__ half_t xs [16 * QSTR];
  __shared__ half_t wls[64 * QSTR];

  const int m0  = blockIdx.x * 16;
  const int n0  = blockIdx.y * 64;
  const int mat = blockIdx.z;
  const float* W = (mat == 0) ? Wq : (mat == 1) ? Wk : Wv;

  const int tid = threadIdx.x;

  // Stage x tile (16 x 384) fp32 -> f16, coalesced float4 loads.
#pragma unroll 4
  for (int i = tid; i < 16 * Dm / 4; i += 128) {
    int r = (i * 4) / Dm, c = (i * 4) % Dm;
    v4f f = *(const v4f*)(x + (size_t)(m0 + r) * Dm + c);
    v4h h = { (_Float16)f.x, (_Float16)f.y, (_Float16)f.z, (_Float16)f.w };
    *(v4h*)(&xs[r * QSTR + c]) = h;
  }
  // Stage W tile (64 x 384) fp32 -> f16.
#pragma unroll 4
  for (int i = tid; i < 64 * Dm / 4; i += 128) {
    int r = (i * 4) / Dm, c = (i * 4) % Dm;
    v4f f = *(const v4f*)(W + (size_t)(n0 + r) * Dm + c);
    v4h h = { (_Float16)f.x, (_Float16)f.y, (_Float16)f.z, (_Float16)f.w };
    *(v4h*)(&wls[r * QSTR + c]) = h;
  }
  __syncthreads();

  const int wave = tid >> 5;
  const int lane = tid & 31;
  const int g    = lane >> 4;   // half-wave (ISA A/B layout group)
  const int ln   = lane & 15;

  v8f c = {};
#pragma unroll
  for (int kk = 0; kk < Dm; kk += 32) {
    // A: row m=ln, k = g*16 + 0..15 contiguous
    v16h a = make_frag(&xs[ln * QSTR + kk + g * 16],
                       &xs[ln * QSTR + kk + g * 16 + 8]);
    // B: column n = W-row (wave*16+ln), k = g*16 + 0..15 contiguous
    const int wrow = wave * 16 + ln;
    v16h b = make_frag(&wls[wrow * QSTR + kk + g * 16],
                       &wls[wrow * QSTR + kk + g * 16 + 8]);
    c = wmma_f16(a, b, c);
  }

  // C layout: VGPR r holds element (m = r + 8*g, n = ln).
  if (mat < 2) {
    half_t* O = (mat == 0) ? Qh : Kh;
#pragma unroll
    for (int r = 0; r < 8; ++r) {
      int mm = m0 + r + g * 8;
      O[(size_t)mm * Dm + n0 + wave * 16 + ln] = (half_t)c[r];
    }
  } else {  // V: store transposed Vt[b][dim][s]
#pragma unroll
    for (int r = 0; r < 8; ++r) {
      int mm = m0 + r + g * 8;
      int bb = mm >> 11, ss = mm & (Sq - 1);
      int dd = n0 + wave * 16 + ln;
      Vt[(size_t)bb * Dm * Sq + (size_t)dd * Sq + ss] = (half_t)c[r];
    }
  }
}

// ---------------------------------------------------------------------------
// Kernel 2: flash attention.  One 16-row query block per WG, 4 waves.
// Wave w owns dims [96w, 96w+96): computes partial scores over its slice
// (LDS cross-wave reduction), redundant online softmax, and accumulates the
// 96 output columns it owns (6 WMMA N-tiles, 48 acc VGPRs).
// ---------------------------------------------------------------------------
__global__ __launch_bounds__(128)
void flash_attn_kernel(const half_t* __restrict__ Qh,
                       const half_t* __restrict__ Kh,
                       const half_t* __restrict__ Vt,
                       float* __restrict__ out) {
  __shared__ half_t qs[16 * QSTR];
  __shared__ float  scred[4][2][16][16];   // per-wave partial 16x32 scores
  __shared__ half_t pbuf[4][16 * PSTR];    // per-wave P tile (f16) for relayout

  const int bid = blockIdx.x;
  const int b   = bid >> 7;            // / (S/16)
  const int q0  = (bid & 127) * 16;
  const size_t rowbase = (size_t)b * Sq;

  const int tid = threadIdx.x;
  // Stage Q block 16x384 (f16), coalesced b128.
#pragma unroll 2
  for (int i = tid; i < 16 * Dm / 8; i += 128) {
    int r = (i * 8) / Dm, c = (i * 8) % Dm;
    *(v8h*)&qs[r * QSTR + c] = *(const v8h*)(Qh + (rowbase + q0 + r) * Dm + c);
  }
  __syncthreads();

  const int wave = tid >> 5, lane = tid & 31, g = lane >> 4, ln = lane & 15;
  const int d0 = wave * 96;

  // Q fragments for this wave's 3 k-steps stay resident (24 VGPRs).
  v16h qf[3];
#pragma unroll
  for (int j = 0; j < 3; ++j)
    qf[j] = make_frag(&qs[ln * QSTR + d0 + j * 32 + g * 16],
                      &qs[ln * QSTR + d0 + j * 32 + g * 16 + 8]);

  v8f acc[6];
#pragma unroll
  for (int nt = 0; nt < 6; ++nt) acc[nt] = (v8f){};
  float mrow[8], lrow[8];
#pragma unroll
  for (int r = 0; r < 8; ++r) { mrow[r] = -__builtin_inff(); lrow[r] = 0.f; }

  const half_t* Kb = Kh + rowbase * Dm;
  const half_t* Vb = Vt + (size_t)b * Dm * Sq;
  const float RLOG2E = 1.44269504088896340736f;

  for (int kb = 0; kb < Sq; kb += 32) {
    // ---- partial scores S = Q K^T over this wave's 96 dims (6 WMMAs) ----
    v8f s0 = {}, s1 = {};
#pragma unroll
    for (int j = 0; j < 3; ++j) {
      const half_t* k0p = Kb + (size_t)(kb + ln) * Dm + d0 + j * 32 + g * 16;
      s0 = wmma_f16(qf[j], make_frag(k0p, k0p + 8), s0);
      const half_t* k1p = Kb + (size_t)(kb + 16 + ln) * Dm + d0 + j * 32 + g * 16;
      s1 = wmma_f16(qf[j], make_frag(k1p, k1p + 8), s1);
    }

    // ---- cross-wave reduction of partials through LDS ----
#pragma unroll
    for (int r = 0; r < 8; ++r) {
      scred[wave][0][r + g * 8][ln] = s0[r];
      scred[wave][1][r + g * 8][ln] = s1[r];
    }
    __syncthreads();
#pragma unroll
    for (int r = 0; r < 8; ++r) {
      float a0 = 0.f, a1 = 0.f;
#pragma unroll
      for (int w = 0; w < 4; ++w) {      // fixed order => identical across waves
        a0 += scred[w][0][r + g * 8][ln];
        a1 += scred[w][1][r + g * 8][ln];
      }
      s0[r] = a0; s1[r] = a1;
    }

    // ---- online softmax (redundant per wave; rows live in 16-lane groups) ----
    float corr[8];
#pragma unroll
    for (int r = 0; r < 8; ++r) {
      float t = fmaxf(s0[r], s1[r]);
      t = fmaxf(t, __shfl_xor(t, 1));
      t = fmaxf(t, __shfl_xor(t, 2));
      t = fmaxf(t, __shfl_xor(t, 4));
      t = fmaxf(t, __shfl_xor(t, 8));
      float mnew = fmaxf(mrow[r], t);
      corr[r] = exp2f((mrow[r] - mnew) * RLOG2E);
      mrow[r] = mnew;
      float p0 = exp2f((s0[r] - mnew) * RLOG2E);
      float p1 = exp2f((s1[r] - mnew) * RLOG2E);
      s0[r] = p0; s1[r] = p1;
      float rs = p0 + p1;
      rs += __shfl_xor(rs, 1);
      rs += __shfl_xor(rs, 2);
      rs += __shfl_xor(rs, 4);
      rs += __shfl_xor(rs, 8);
      lrow[r] = lrow[r] * corr[r] + rs;
    }
#pragma unroll
    for (int nt = 0; nt < 6; ++nt)
#pragma unroll
      for (int r = 0; r < 8; ++r)
        acc[nt][r] *= corr[r];

    // ---- P (16x32, f16) to LDS for C-layout -> A-layout relayout ----
    half_t* pb = &pbuf[wave][0];
#pragma unroll
    for (int r = 0; r < 8; ++r) {
      pb[(r + g * 8) * PSTR + ln]      = (half_t)s0[r];
      pb[(r + g * 8) * PSTR + 16 + ln] = (half_t)s1[r];
    }
    __syncthreads();   // also protects scred reuse next iteration

    // ---- PV: A = P fragment, B = Vt rows (contiguous keys) ----
    v16h pf = make_frag(&pb[ln * PSTR + g * 16], &pb[ln * PSTR + g * 16 + 8]);
#pragma unroll
    for (int nt = 0; nt < 6; ++nt) {
      const half_t* vp = Vb + (size_t)(d0 + nt * 16 + ln) * Sq + kb + g * 16;
      acc[nt] = wmma_f16(pf, make_frag(vp, vp + 8), acc[nt]);
    }

    // gfx1250 global_prefetch_b8 for next K/V tiles
    if (kb + 32 < Sq) {
      __builtin_prefetch(Kb + (size_t)(kb + 32 + ln) * Dm + d0, 0, 1);
      __builtin_prefetch(Vb + (size_t)(d0 + (lane % 6) * 16) * Sq + kb + 32, 0, 1);
    }
  }

  // ---- finalize: divide by softmax denominator, store f32 ----
#pragma unroll
  for (int r = 0; r < 8; ++r) lrow[r] = 1.0f / lrow[r];
#pragma unroll
  for (int nt = 0; nt < 6; ++nt)
#pragma unroll
    for (int r = 0; r < 8; ++r) {
      int mm = r + g * 8;
      out[((size_t)b * Sq + q0 + mm) * Dm + d0 + nt * 16 + ln] =
          acc[nt][r] * lrow[r];
    }
}

// ---------------------------------------------------------------------------
extern "C" void kernel_launch(void* const* d_in, const int* in_sizes, int n_in,
                              void* d_out, int out_size, void* d_ws, size_t ws_size,
                              hipStream_t stream) {
  (void)in_sizes; (void)n_in; (void)out_size; (void)ws_size;
  const float* x  = (const float*)d_in[0];
  const float* Wq = (const float*)d_in[1];
  const float* Wk = (const float*)d_in[2];
  const float* Wv = (const float*)d_in[3];
  float* out = (float*)d_out;

  // Workspace: Qh | Kh | Vt  (f16), 3 * 16384 * 384 * 2 B = 37.75 MB
  half_t* Qh = (half_t*)d_ws;
  half_t* Kh = Qh + (size_t)Mrow * Dm;
  half_t* Vt = Kh + (size_t)Mrow * Dm;

  dim3 gp(Mrow / 16, Dm / 64, 3);          // 1024 x 6 x 3 workgroups
  qkv_proj_kernel<<<gp, 128, 0, stream>>>(x, Wq, Wk, Wv, Qh, Kh, Vt);

  flash_attn_kernel<<<dim3(Bb * (Sq / 16)), 128, 0, stream>>>(Qh, Kh, Vt, out);
}
